// CrossAttention_78864189489505
// MI455X (gfx1250) — compile-verified, matrix operand order checked by reading, not run on hardware
//
#include <hip/hip_runtime.h>
#include <hip/hip_bf16.h>
#include <math.h>

typedef _Float16 v16h __attribute__((ext_vector_type(16)));
typedef _Float16 v8h  __attribute__((ext_vector_type(8)));
typedef float    v8f  __attribute__((ext_vector_type(8)));

#define B_  16
#define LQ_ 2048
#define LK_ 2048
#define DQ_ 768
#define DK_ 1024

__device__ __forceinline__ v8f splat8(float x) {
  v8f r;
#pragma unroll
  for (int i = 0; i < 8; ++i) r[i] = x;
  return r;
}

// Build a 16x32 f16 A/B fragment register for this lane from a row base ptr.
// ISA layout: lanes 0-15 hold K 0..7 / 16..23, lanes 16-31 hold K 8..15 / 24..31.
__device__ __forceinline__ v16h load_frag(const _Float16* rowbase, int half) {
  const _Float16* p = rowbase + (half ? 8 : 0);
  v8h lo = *(const v8h*)(p);
  v8h hi = *(const v8h*)(p + 16);
  return __builtin_shufflevector(lo, hi, 0, 1, 2, 3, 4, 5, 6, 7,
                                 8, 9, 10, 11, 12, 13, 14, 15);
}

__device__ __forceinline__ v8f wmma_f16(v16h a, v16h b, v8f c) {
  return __builtin_amdgcn_wmma_f32_16x16x32_f16(false, a, false, b,
                                                (short)0, c, false, false);
}

// ---------------------------------------------------------------------------
// Kernel 1: W [D][N] f32  ->  Wt [N][D] f16   (so B-operand rows are N-major)
// ---------------------------------------------------------------------------
__global__ __launch_bounds__(256) void wt_transpose_f16(
    const float* __restrict__ W, _Float16* __restrict__ Wt, int D, int N) {
  __shared__ float tile[16][17];
  const int n0 = blockIdx.x * 16, d0 = blockIdx.y * 16;
  const int tx = threadIdx.x, ty = threadIdx.y;
  tile[ty][tx] = W[(size_t)(d0 + ty) * N + (n0 + tx)];
  __syncthreads();
  Wt[(size_t)(n0 + ty) * D + (d0 + tx)] = (_Float16)tile[tx][ty];
}

// ---------------------------------------------------------------------------
// Kernel 2: projection GEMM  out[m][n] = X[m][:] . W[:][n] + bias[n]
//   X  : f32 [M][Kd]          (activations, converted to f16 on the fly)
//   Wt : f16 [1024][Kd]       (pre-transposed weights = Bt rows)
//   out_rm : f16 [M][1024]            (row-major, for q / k)
//   out_tr : f16 [B][1024][2048]      (dim-major, for v -> vT)
// Block tile 64(M) x 128(N), 8 waves = 2x4 grid of 32x32 wave tiles.
// ---------------------------------------------------------------------------
__global__ __launch_bounds__(256) void proj_gemm(
    const float* __restrict__ X, const _Float16* __restrict__ Wt,
    const float* __restrict__ bias, _Float16* __restrict__ out_rm,
    _Float16* __restrict__ out_tr, int Kd) {
  __shared__ _Float16 As[64][40];    // padded: 80B row stride, b128-safe
  __shared__ _Float16 Bs[128][40];

  const int t = threadIdx.x;
  const int wave = t >> 5, lane = t & 31;
  const int hh = lane >> 4, ln = lane & 15;
  const int wm = wave & 1, wn = wave >> 1;
  const int m_blk = blockIdx.y * 64;
  const int n_blk = blockIdx.x * 128;

  v8f acc[2][2];
#pragma unroll
  for (int j = 0; j < 2; ++j) {
    float bv = bias[n_blk + wn * 32 + j * 16 + ln];
    acc[0][j] = splat8(bv);
    acc[1][j] = splat8(bv);
  }

  const int arow = t >> 2, ac8 = (t & 3) * 8;   // 64 rows x 32 f32
  const int brow = t >> 1, bc16 = (t & 1) * 16; // 128 rows x 32 f16

  const float*    aptr = X + (size_t)(m_blk + arow) * Kd + ac8;
  const _Float16* bptr = Wt + (size_t)(n_blk + brow) * Kd + bc16;

  for (int k0 = 0; k0 < Kd; k0 += 32) {
    const float* ap = aptr + k0;
    float4 f0 = *(const float4*)ap;
    float4 f1 = *(const float4*)(ap + 4);
    v8h ah;
    ah[0] = (_Float16)f0.x; ah[1] = (_Float16)f0.y;
    ah[2] = (_Float16)f0.z; ah[3] = (_Float16)f0.w;
    ah[4] = (_Float16)f1.x; ah[5] = (_Float16)f1.y;
    ah[6] = (_Float16)f1.z; ah[7] = (_Float16)f1.w;
    *(v8h*)&As[arow][ac8] = ah;

    const _Float16* bp = bptr + k0;
    *(v8h*)&Bs[brow][bc16]     = *(const v8h*)bp;
    *(v8h*)&Bs[brow][bc16 + 8] = *(const v8h*)(bp + 8);

    // pull the next K-step's tiles toward L0/L2 while we compute
    if (k0 + 32 < Kd) {
      __builtin_prefetch(ap + 32, 0, 3);
      __builtin_prefetch(bp + 32, 0, 3);
    }
    __syncthreads();

    v16h af0 = load_frag(&As[wm * 32 + ln][0], hh);
    v16h af1 = load_frag(&As[wm * 32 + 16 + ln][0], hh);
    v16h bf0 = load_frag(&Bs[wn * 32 + ln][0], hh);
    v16h bf1 = load_frag(&Bs[wn * 32 + 16 + ln][0], hh);
    acc[0][0] = wmma_f16(af0, bf0, acc[0][0]);
    acc[0][1] = wmma_f16(af0, bf1, acc[0][1]);
    acc[1][0] = wmma_f16(af1, bf0, acc[1][0]);
    acc[1][1] = wmma_f16(af1, bf1, acc[1][1]);
    __syncthreads();
  }

#pragma unroll
  for (int i = 0; i < 2; ++i) {
#pragma unroll
    for (int j = 0; j < 2; ++j) {
      const int n = n_blk + wn * 32 + j * 16 + ln;
      const int m0 = m_blk + wm * 32 + i * 16 + hh * 8; // D-frag: M = r + 8*hh
      if (out_rm) {
#pragma unroll
        for (int r = 0; r < 8; ++r)
          out_rm[(size_t)(m0 + r) * 1024 + n] = (_Float16)acc[i][j][r];
      }
      if (out_tr) {
        const int batch = m0 >> 11, tok = m0 & 2047;
        v8h h;
#pragma unroll
        for (int r = 0; r < 8; ++r) h[r] = (_Float16)acc[i][j][r];
        *(v8h*)&out_tr[((size_t)batch * 1024 + n) * LK_ + tok] = h;
      }
    }
  }
}

// ---------------------------------------------------------------------------
// Kernel 3: flash attention. One block = (batch, 16 queries). 8 waves.
//  Per 128-key iteration: wave w computes S tile for keys [kb+16w, +16)
//  (32 WMMAs over 1024 dims), online softmax across waves via LDS, then
//  wave w accumulates its private 128-dim slice of O (32 WMMAs vs vT).
// ---------------------------------------------------------------------------
__global__ __launch_bounds__(256) void attention_kernel(
    const _Float16* __restrict__ qh, const _Float16* __restrict__ kh,
    const _Float16* __restrict__ vT, float* __restrict__ out) {
  __shared__ _Float16 Qs[16][1032];  // stride 2064B -> 4-bank rotation
  __shared__ _Float16 Ps[16][136];
  __shared__ float wmaxs[8][16];
  __shared__ float wsums[8][16];

  const int t = threadIdx.x;
  const int wave = t >> 5, lane = t & 31;
  const int hh = lane >> 4, ln = lane & 15;
  const int q0 = blockIdx.x * 16;
  const int batch = blockIdx.y;

  {  // stage the 16x1024 f16 Q tile
    const int row = t >> 4, seg = t & 15;
    const _Float16* src = qh + ((size_t)batch * LQ_ + q0 + row) * 1024 + seg * 64;
#pragma unroll
    for (int u = 0; u < 8; ++u)
      *(v8h*)&Qs[row][seg * 64 + u * 8] = *(const v8h*)(src + u * 8);
  }
  __syncthreads();

  v8f o[8];
#pragma unroll
  for (int i = 0; i < 8; ++i) o[i] = splat8(0.f);
  v8f m_run = splat8(-__builtin_inff());
  v8f l_run = splat8(0.f);

  const _Float16* kbatch = kh + (size_t)batch * LK_ * 1024;
  const _Float16* vbatch = vT + (size_t)batch * 1024 * LK_;

  for (int kb = 0; kb < LK_; kb += 128) {
    // ---- S = (Q K^T) * 1/sqrt(1024) for this wave's 16 keys
    v8f s = splat8(0.f);
    const _Float16* krow = kbatch + (size_t)(kb + wave * 16 + ln) * 1024;
    if (kb + 128 < LK_) {
      // prefetch next key block's row for this lane (cacheline granular)
      __builtin_prefetch(krow + (size_t)128 * 1024, 0, 3);
      __builtin_prefetch(krow + (size_t)128 * 1024 + 512, 0, 3);
    }
#pragma unroll 4
    for (int d0 = 0; d0 < 1024; d0 += 32) {
      v16h aq = load_frag(&Qs[ln][d0], hh);
      v16h bk = load_frag(krow + d0, hh);
      s = wmma_f16(aq, bk, s);
    }
#pragma unroll
    for (int r = 0; r < 8; ++r) s[r] *= 0.03125f;

    // ---- per-row max of this wave's tile (reduce across 16 lanes/half)
    v8f tmax;
#pragma unroll
    for (int r = 0; r < 8; ++r) {
      float x = s[r];
      x = fmaxf(x, __shfl_xor(x, 1, 32));
      x = fmaxf(x, __shfl_xor(x, 2, 32));
      x = fmaxf(x, __shfl_xor(x, 4, 32));
      x = fmaxf(x, __shfl_xor(x, 8, 32));
      tmax[r] = x;
    }
    if (ln == 0) {
#pragma unroll
      for (int r = 0; r < 8; ++r) wmaxs[wave][hh * 8 + r] = tmax[r];
    }
    __syncthreads();

    v8f m_new, alpha;
#pragma unroll
    for (int r = 0; r < 8; ++r) {
      float mx = m_run[r];
#pragma unroll
      for (int ww = 0; ww < 8; ++ww) mx = fmaxf(mx, wmaxs[ww][hh * 8 + r]);
      m_new[r] = mx;
      alpha[r] = __expf(m_run[r] - mx);
    }

    // ---- P = exp(S - m_new); stash in LDS for the PV GEMM; partial sums
    v8f p;
#pragma unroll
    for (int r = 0; r < 8; ++r) {
      p[r] = __expf(s[r] - m_new[r]);
      Ps[hh * 8 + r][wave * 16 + ln] = (_Float16)p[r];
    }
    v8f tsum;
#pragma unroll
    for (int r = 0; r < 8; ++r) {
      float x = p[r];
      x += __shfl_xor(x, 1, 32);
      x += __shfl_xor(x, 2, 32);
      x += __shfl_xor(x, 4, 32);
      x += __shfl_xor(x, 8, 32);
      tsum[r] = x;
    }
    if (ln == 0) {
#pragma unroll
      for (int r = 0; r < 8; ++r) wsums[wave][hh * 8 + r] = tsum[r];
    }
    __syncthreads();

#pragma unroll
    for (int r = 0; r < 8; ++r) {
      float su = 0.f;
#pragma unroll
      for (int ww = 0; ww < 8; ++ww) su += wsums[ww][hh * 8 + r];
      l_run[r] = alpha[r] * l_run[r] + su;
      m_run[r] = m_new[r];
    }

    // ---- O = alpha*O + P.V for this wave's 128 output dims
#pragma unroll
    for (int dt = 0; dt < 8; ++dt) {
      const _Float16* vrow =
          vbatch + (size_t)(wave * 128 + dt * 16 + ln) * LK_ + kb;
      if (kb + 128 < LK_) __builtin_prefetch(vrow + 128, 0, 3);
      v8f oo;
#pragma unroll
      for (int r = 0; r < 8; ++r) oo[r] = o[dt][r] * alpha[r];
#pragma unroll
      for (int kk = 0; kk < 128; kk += 32) {
        v16h ap = load_frag(&Ps[ln][kk], hh);
        v16h bv = load_frag(vrow + kk, hh);
        oo = wmma_f16(ap, bv, oo);
      }
      o[dt] = oo;
    }
    __syncthreads();  // protect Ps / wmaxs / wsums for next iteration
  }

  v8f inv;
#pragma unroll
  for (int r = 0; r < 8; ++r) inv[r] = 1.f / l_run[r];
#pragma unroll
  for (int dt = 0; dt < 8; ++dt) {
    const int n = wave * 128 + dt * 16 + ln;
#pragma unroll
    for (int r = 0; r < 8; ++r) {
      out[((size_t)batch * LQ_ + q0 + hh * 8 + r) * 1024 + n] =
          o[dt][r] * inv[r];
    }
  }
}

// ---------------------------------------------------------------------------
extern "C" void kernel_launch(void* const* d_in, const int* in_sizes, int n_in,
                              void* d_out, int out_size, void* d_ws, size_t ws_size,
                              hipStream_t stream) {
  const float* query = (const float*)d_in[0];  // [16,2048,768]
  const float* key   = (const float*)d_in[1];  // [16,2048,1024]
  const float* Wq    = (const float*)d_in[2];  // [768,1024]
  const float* bq    = (const float*)d_in[3];
  const float* Wk    = (const float*)d_in[4];  // [1024,1024]
  const float* bk    = (const float*)d_in[5];
  const float* Wv    = (const float*)d_in[6];  // [1024,1024]
  const float* bv    = (const float*)d_in[7];
  float* out = (float*)d_out;

  // workspace layout (f16)
  _Float16* qh  = (_Float16*)d_ws;                       // 16*2048*1024
  _Float16* kh  = qh + (size_t)B_ * LQ_ * 1024;
  _Float16* vT  = kh + (size_t)B_ * LK_ * 1024;          // [B][1024][2048]
  _Float16* WqT = vT + (size_t)B_ * 1024 * LK_;          // [1024][768]
  _Float16* WkT = WqT + (size_t)1024 * DQ_;              // [1024][1024]
  _Float16* WvT = WkT + (size_t)1024 * DK_;              // [1024][1024]

  dim3 tb(16, 16);
  wt_transpose_f16<<<dim3(1024 / 16, DQ_ / 16), tb, 0, stream>>>(Wq, WqT, DQ_, 1024);
  wt_transpose_f16<<<dim3(1024 / 16, DK_ / 16), tb, 0, stream>>>(Wk, WkT, DK_, 1024);
  wt_transpose_f16<<<dim3(1024 / 16, DK_ / 16), tb, 0, stream>>>(Wv, WvT, DK_, 1024);

  const int M = B_ * LQ_;  // 32768
  dim3 pg(1024 / 128, M / 64);
  proj_gemm<<<pg, 256, 0, stream>>>(query, WqT, bq, qh, nullptr, DQ_);
  proj_gemm<<<pg, 256, 0, stream>>>(key,   WkT, bk, kh, nullptr, DK_);
  proj_gemm<<<pg, 256, 0, stream>>>(key,   WvT, bv, nullptr, vT, DK_);

  attention_kernel<<<dim3(LQ_ / 16, B_), 256, 0, stream>>>(qh, kh, vT, out);
}
